// GatedElman_738734375257
// MI455X (gfx1250) — matrix operand order repaired and checked by reading, not compiled, not run on hardware
//
#include <hip/hip_runtime.h>
#include <cstdint>

#define DIM    1024
#define TLEN   1024
#define BATCH  4
#define NDEPTH 4
#define VOCAB  32000
#define ROWS   (BATCH * TLEN)   // 4096

typedef __attribute__((ext_vector_type(16))) __bf16 bf16x16;
typedef __attribute__((ext_vector_type(8)))  __bf16 bf16x8;
typedef __attribute__((ext_vector_type(8)))  float  f32x8;
typedef __attribute__((ext_vector_type(4)))  int    i32x4;

// gfx1250 async global->LDS path (ASYNCcnt); guarded so compile never breaks.
#if defined(__gfx1250__) && __has_builtin(__builtin_amdgcn_global_load_async_to_lds_b128)
#define HAVE_ASYNC_LDS 1
#else
#define HAVE_ASYNC_LDS 0
#endif

__device__ __forceinline__ void async_cp16(const __bf16* g, __bf16* l) {
#if HAVE_ASYNC_LDS
  // Builtin signature (from hipcc diagnostic): param0 = int4 AS(1)*, then LDS
  // dest AS(3)*, imm offset, imm cpol. Emits global_load_async_to_lds_b128.
  __builtin_amdgcn_global_load_async_to_lds_b128(
      (__attribute__((address_space(1))) i32x4*)(g),
      (__attribute__((address_space(3))) i32x4*)(l), /*offset=*/0, /*cpol=*/0);
#else
  *(uint4*)l = *(const uint4*)g;  // synchronous fallback (still LDS-staged)
#endif
}

template <int N>
__device__ __forceinline__ void wait_async() {
#if HAVE_ASYNC_LDS
#if __has_builtin(__builtin_amdgcn_s_wait_asynccnt)
  __builtin_amdgcn_s_wait_asynccnt(N);
#else
  asm volatile("s_wait_asynccnt %0" ::"i"(N) : "memory");
#endif
#endif
}

// ---------------------------------------------------------------------------
// WMMA helpers (gfx1250 wave32, V_WMMA_F32_16X16X32_BF16)
// A fragment (16x32 bf16, MxK): lane m = lane&15, half = lane>>4.
//   VGPR0..3 hold K = 8*half .. +7, VGPR4..7 hold K = 16+8*half .. +7.
// B fragment (32x16 bf16, KxN): lane n = lane&15; same K order as A (8*half.., 16+8*half..).
// C/D (16x16 f32): lane col = lane&15; rows rbase..rbase+7, rbase = half*8.
// ---------------------------------------------------------------------------
__device__ __forceinline__ bf16x16 ldA(const __bf16* base, int half) {
  union { bf16x16 v; struct { bf16x8 lo; bf16x8 hi; } p; } f;
  f.p.lo = *(const bf16x8*)(base + 8 * half);
  f.p.hi = *(const bf16x8*)(base + 16 + 8 * half);
  return f.v;
}
__device__ __forceinline__ bf16x16 ldB(const __bf16* base, int half) {
  union { bf16x16 v; struct { bf16x8 lo; bf16x8 hi; } p; } f;
  f.p.lo = *(const bf16x8*)(base + 8 * half);
  f.p.hi = *(const bf16x8*)(base + 16 + 8 * half);
  return f.v;
}
__device__ __forceinline__ f32x8 wmma_bf16(bf16x16 a, bf16x16 b, f32x8 c) {
  return __builtin_amdgcn_wmma_f32_16x16x32_bf16(
      /*neg_a=*/false, a, /*neg_b=*/false, b,
      /*c_mod=*/(short)0, c, /*reuse_a=*/false, /*reuse_b=*/false);
}

// ---------------------------------------------------------------------------
// Embedding gather: X[row,:] = embed[idx[row],:]
// ---------------------------------------------------------------------------
__global__ __launch_bounds__(256) void gather_rows(const int* __restrict__ idx,
                                                   const float* __restrict__ emb,
                                                   float* __restrict__ out) {
  const int row = blockIdx.x;
  const int id = idx[row];
  const float4* src = (const float4*)(emb + (size_t)id * DIM);
  float4* dst = (float4*)(out + (size_t)row * DIM);
  for (int i = threadIdx.x; i < DIM / 4; i += 256) dst[i] = src[i];
}

// fp32 -> bf16 flat convert (embedding table copy for the tied head)
__global__ __launch_bounds__(256) void cvt_bf16(const float* __restrict__ in,
                                                __bf16* __restrict__ out, size_t n) {
  size_t i = (size_t)blockIdx.x * 256 + threadIdx.x;
  if (i < n) out[i] = (__bf16)in[i];
}

// fp32 [K][N] -> bf16 [N][K] (LDS-tiled transpose + convert)
__global__ __launch_bounds__(256) void transpose_cvt(const float* __restrict__ in,
                                                     __bf16* __restrict__ out,
                                                     int K, int N) {
  __shared__ float tile[32][33];
  const int tx = threadIdx.x & 31;
  const int ty = threadIdx.x >> 5;
  const int nbase = blockIdx.x * 32;
  const int kbase = blockIdx.y * 32;
#pragma unroll
  for (int i = 0; i < 32; i += 8)
    tile[ty + i][tx] = in[(size_t)(kbase + ty + i) * N + nbase + tx];
  __syncthreads();
#pragma unroll
  for (int i = 0; i < 32; i += 8)
    out[(size_t)(nbase + ty + i) * K + kbase + tx] = (__bf16)tile[tx][ty + i];
}

// ---------------------------------------------------------------------------
// Row LayerNorm (D=1024) -> bf16 output for WMMA A operands
// ---------------------------------------------------------------------------
__global__ __launch_bounds__(256) void layernorm_bf16(const float* __restrict__ X,
                                                      const float* __restrict__ g,
                                                      const float* __restrict__ b,
                                                      __bf16* __restrict__ Y) {
  const int row = blockIdx.x;
  const float* x = X + (size_t)row * DIM;
  float s = 0.f, ss = 0.f;
  for (int i = threadIdx.x; i < DIM; i += 256) {
    float v = x[i];
    s += v; ss += v * v;
  }
  __shared__ float red[2][8];
  const int lane = threadIdx.x & 31, w = threadIdx.x >> 5;
#pragma unroll
  for (int o = 16; o > 0; o >>= 1) {
    s += __shfl_down(s, o, 32);
    ss += __shfl_down(ss, o, 32);
  }
  if (lane == 0) { red[0][w] = s; red[1][w] = ss; }
  __syncthreads();
  if (w == 0) {
    s = (lane < 8) ? red[0][lane] : 0.f;
    ss = (lane < 8) ? red[1][lane] : 0.f;
#pragma unroll
    for (int o = 4; o > 0; o >>= 1) {
      s += __shfl_down(s, o, 32);
      ss += __shfl_down(ss, o, 32);
    }
    if (lane == 0) { red[0][0] = s; red[1][0] = ss; }
  }
  __syncthreads();
  const float mu = red[0][0] * (1.f / DIM);
  const float var = red[1][0] * (1.f / DIM) - mu * mu;
  const float rinv = rsqrtf(var + 1e-5f);
  for (int i = threadIdx.x; i < DIM; i += 256)
    Y[(size_t)row * DIM + i] = (__bf16)((x[i] - mu) * rinv * g[i] + b[i]);
}

// ---------------------------------------------------------------------------
// bf16 WMMA GEMM: C[M,N](f32) = A[M,K](bf16) @ Bt[N,K](bf16, pre-transposed)
// Block = 8 waves, macro-tile 128(M) x 64(N), K-step 32.
// A/B K-slices staged in LDS via async global->LDS copies, double-buffered:
//   issue stage s^1, s_wait_asynccnt<=3 (stage s landed, s^1 in flight),
//   barrier, WMMA from LDS (ds_load_b128), barrier.
// B slice is shared by all 8 waves -> 8x less L2 traffic than private loads.
// grid = (N/64, M/128)
// ---------------------------------------------------------------------------
__global__ __launch_bounds__(256) void gemm_bf16_kernel(const __bf16* __restrict__ A,
                                                        const __bf16* __restrict__ Bt,
                                                        float* __restrict__ C,
                                                        int N, int K) {
  __shared__ __align__(32) __bf16 Abuf[2][128][32];  // 16 KB
  __shared__ __align__(32) __bf16 Bbuf[2][64][32];   //  8 KB
  const int tid = threadIdx.x;
  const int lane = tid & 31;
  const int wave = tid >> 5;
  const int half = lane >> 4;
  const int l16 = lane & 15;
  const int mblk = blockIdx.y * 128;
  const int n0 = blockIdx.x * 64;

  // Stage copy: 512 A chunks (128 rows x 4x16B) + 256 B chunks (64 rows x 4x16B)
  // = 3 async ops per thread per stage.
  auto copy_stage = [&](int s, int k0) {
#pragma unroll
    for (int c = tid, it = 0; it < 2; c += 256, ++it) {
      const int row = c >> 2, part = c & 3;
      async_cp16(A + (size_t)(mblk + row) * K + k0 + part * 8, &Abuf[s][row][part * 8]);
    }
    {
      const int col = tid >> 2, part = tid & 3;
      async_cp16(Bt + (size_t)(n0 + col) * K + k0 + part * 8, &Bbuf[s][col][part * 8]);
    }
  };

  f32x8 acc0 = {}, acc1 = {}, acc2 = {}, acc3 = {};
  const int nk = K / 32;
  copy_stage(0, 0);
  for (int i = 0; i < nk; ++i) {
    const int s = i & 1;
    const bool more = (i + 1) < nk;
    if (more) copy_stage(s ^ 1, (i + 1) * 32);
    if (more) wait_async<3>(); else wait_async<0>();
    __syncthreads();  // stage s visible to all waves
    bf16x16 a = ldA(&Abuf[s][wave * 16 + l16][0], half);
    acc0 = wmma_bf16(a, ldB(&Bbuf[s][0 + l16][0], half), acc0);
    acc1 = wmma_bf16(a, ldB(&Bbuf[s][16 + l16][0], half), acc1);
    acc2 = wmma_bf16(a, ldB(&Bbuf[s][32 + l16][0], half), acc2);
    acc3 = wmma_bf16(a, ldB(&Bbuf[s][48 + l16][0], half), acc3);
    __syncthreads();  // stage s reads done before it is overwritten
  }

  const int m0 = mblk + wave * 16;
  const int rbase = half ? 8 : 0;
  f32x8 accs[4] = {acc0, acc1, acc2, acc3};
#pragma unroll
  for (int j = 0; j < 4; ++j)
#pragma unroll
    for (int r = 0; r < 8; ++r)
      C[(size_t)(m0 + rbase + r) * N + (n0 + 16 * j + l16)] = accs[j][r];
}

// ---------------------------------------------------------------------------
// Persistent Elman recurrence (one workgroup, 32 waves, hidden state in LDS).
// h is [BATCH,DIM] padded to 16 rows (rows 4..15 stay exactly 0 through tanh).
// Per step: h_new = tanh(wx_t + h @ Wh); X += sig(zh)*h_new + sig(zx)*wx_t.
// Each wave owns 2 of the 64 output column tiles; 2 barriers per step.
// ---------------------------------------------------------------------------
__global__ __launch_bounds__(1024) void recurrence_kernel(float* __restrict__ X,
                                                          const float* __restrict__ WXT,
                                                          const float* __restrict__ Z,
                                                          const __bf16* __restrict__ WhT) {
  __shared__ __align__(32) __bf16 hbuf[16][DIM];  // 32 KB
  const int tid = threadIdx.x;
  const int lane = tid & 31, wave = tid >> 5;
  const int half = lane >> 4, l16 = lane & 15;
  const int rbase = half ? 8 : 0;
  for (int i = tid; i < 16 * DIM; i += 1024) ((__bf16*)hbuf)[i] = (__bf16)0.f;
  __syncthreads();

  const int n0a = (2 * wave + 0) * 16;
  const int n0b = (2 * wave + 1) * 16;
  const __bf16* a_row = &hbuf[l16][0];
  const __bf16* brow_a = WhT + (size_t)(n0a + l16) * DIM;
  const __bf16* brow_b = WhT + (size_t)(n0b + l16) * DIM;

  for (int t = 0; t < TLEN; ++t) {
    f32x8 acc_a = {}, acc_b = {};
    for (int k0 = 0; k0 < DIM; k0 += 32) {
      bf16x16 a = ldA(a_row + k0, half);  // ds_load_b128 from LDS-resident state
      acc_a = wmma_bf16(a, ldB(brow_a + k0, half), acc_a);
      acc_b = wmma_bf16(a, ldB(brow_b + k0, half), acc_b);
    }
    __syncthreads();  // all reads of hbuf done before anyone overwrites it
#pragma unroll
    for (int jj = 0; jj < 2; ++jj) {
      const int col = (jj ? n0b : n0a) + l16;
      f32x8 acc = jj ? acc_b : acc_a;
#pragma unroll
      for (int r = 0; r < 8; ++r) {
        const int m = rbase + r;
        float wx = 0.f;
        if (m < BATCH) wx = WXT[((size_t)m * TLEN + t) * DIM + col];
        const float hn = tanhf(wx + acc[r]);
        hbuf[m][col] = (__bf16)hn;
        if (m < BATCH) {
          const float* zrow = Z + ((size_t)m * TLEN + t) * (2 * DIM);
          const float gh = 1.f / (1.f + expf(-zrow[col]));
          const float gx = 1.f / (1.f + expf(-zrow[DIM + col]));
          const size_t xi = ((size_t)m * TLEN + t) * DIM + col;
          X[xi] += gh * hn + gx * wx;  // fused gated output + residual add
        }
      }
    }
    __syncthreads();  // writes visible before next step's reads
  }
}

// ---------------------------------------------------------------------------
// Host orchestration (graph-capture safe: no alloc, everything on stream)
// ---------------------------------------------------------------------------
extern "C" void kernel_launch(void* const* d_in, const int* in_sizes, int n_in,
                              void* d_out, int out_size, void* d_ws, size_t ws_size,
                              hipStream_t stream) {
  (void)in_sizes; (void)n_in; (void)out_size; (void)ws_size;
  const int*   x      = (const int*)d_in[0];
  const float* embed  = (const float*)d_in[1];
  const float* Wx     = (const float*)d_in[2];
  const float* Wh     = (const float*)d_in[3];
  const float* Wz     = (const float*)d_in[4];
  const float* ln_g   = (const float*)d_in[5];
  const float* ln_b   = (const float*)d_in[6];
  const float* norm_g = (const float*)d_in[7];
  const float* norm_b = (const float*)d_in[8];
  float* logits = (float*)d_out;

  // Carve scratch (~150 MB total)
  char* ws = (char*)d_ws;
  auto carve = [&](size_t bytes) -> char* {
    char* p = ws;
    ws += (bytes + 255) & ~(size_t)255;
    return p;
  };
  float*  X    = (float*) carve((size_t)ROWS * DIM * 4);       // hidden stream (fp32)
  __bf16* XNb  = (__bf16*)carve((size_t)ROWS * DIM * 2);       // LN output (bf16 A)
  float*  WXT  = (float*) carve((size_t)ROWS * DIM * 4);       // xn @ Wx
  float*  Zb   = (float*) carve((size_t)ROWS * 2 * DIM * 4);   // xn @ Wz
  __bf16* WxTb = (__bf16*)carve((size_t)DIM * DIM * 2);        // Wx^T bf16
  __bf16* WhTb = (__bf16*)carve((size_t)DIM * DIM * 2);        // Wh^T bf16
  __bf16* WzTb = (__bf16*)carve((size_t)2 * DIM * DIM * 2);    // Wz^T bf16
  __bf16* EMBb = (__bf16*)carve((size_t)VOCAB * DIM * 2);      // embed bf16 (tied head B)

  gather_rows<<<ROWS, 256, 0, stream>>>(x, embed, X);
  {
    const size_t n = (size_t)VOCAB * DIM;
    cvt_bf16<<<(unsigned)((n + 255) / 256), 256, 0, stream>>>(embed, EMBb, n);
  }

  for (int l = 0; l < NDEPTH; ++l) {
    transpose_cvt<<<dim3(DIM / 32, DIM / 32), 256, 0, stream>>>(
        Wx + (size_t)l * DIM * DIM, WxTb, DIM, DIM);
    transpose_cvt<<<dim3(DIM / 32, DIM / 32), 256, 0, stream>>>(
        Wh + (size_t)l * DIM * DIM, WhTb, DIM, DIM);
    transpose_cvt<<<dim3(2 * DIM / 32, DIM / 32), 256, 0, stream>>>(
        Wz + (size_t)l * DIM * 2 * DIM, WzTb, DIM, 2 * DIM);

    layernorm_bf16<<<ROWS, 256, 0, stream>>>(X, ln_g + l * DIM, ln_b + l * DIM, XNb);
    gemm_bf16_kernel<<<dim3(DIM / 64, ROWS / 128), 256, 0, stream>>>(XNb, WxTb, WXT, DIM, DIM);
    gemm_bf16_kernel<<<dim3(2 * DIM / 64, ROWS / 128), 256, 0, stream>>>(XNb, WzTb, Zb, 2 * DIM, DIM);
    recurrence_kernel<<<1, 1024, 0, stream>>>(X, WXT, Zb, WhTb);
  }

  layernorm_bf16<<<ROWS, 256, 0, stream>>>(X, norm_g, norm_b, XNb);
  // Tied head: B^T columns are embed rows -> EMBb used directly, no transpose.
  gemm_bf16_kernel<<<dim3(VOCAB / 64, ROWS / 128), 256, 0, stream>>>(XNb, EMBb, logits, VOCAB, DIM);
}
